// CausalSelfAttention_63376537420223
// MI455X (gfx1250) — matrix operand order repaired and checked by reading, not compiled, not run on hardware
//
#include <hip/hip_runtime.h>

typedef __bf16 bf16_t;
typedef __attribute__((ext_vector_type(16))) __bf16 bf16x16;
typedef __attribute__((ext_vector_type(8)))  __bf16 bf16x8;
typedef __attribute__((ext_vector_type(8)))  float  f32x8;

#define T_SEQ 4096
#define NB    2
#define C_EMB 768
#define NH    12
#define DH    64
#define C3    2304

__device__ __forceinline__ bf16x16 cat8(bf16x8 lo, bf16x8 hi) {
  return __builtin_shufflevector(lo, hi, 0,1,2,3,4,5,6,7,8,9,10,11,12,13,14,15);
}

__device__ __forceinline__ f32x8 wmma_bf16(bf16x16 a, bf16x16 b, f32x8 c) {
  return __builtin_amdgcn_wmma_f32_16x16x32_bf16(false, a, false, b, (short)0, c,
                                                 false, false);
}

__global__ void cvt_bf16_kernel(const float* __restrict__ in, bf16_t* __restrict__ out, int n) {
  int i = blockIdx.x * blockDim.x + threadIdx.x;
  if (i < n) out[i] = (bf16_t)in[i];
}

// W[rows,cols] (f32) -> Wt[cols,rows] (bf16)
__global__ void transpose_cvt_kernel(const float* __restrict__ W, bf16_t* __restrict__ Wt,
                                     int rows, int cols) {
  int i = blockIdx.x * blockDim.x + threadIdx.x;
  if (i < rows * cols) {
    int n = i / rows;
    int k = i - n * rows;
    Wt[i] = (bf16_t)W[(size_t)k * cols + n];
  }
}

// ---------------------------------------------------------------------------
// Register-blocked GEMM mainloop: one wave computes a 32x64 f32 tile of
// A[M,K] * Bt[N,K]^T (bf16, Bt pre-transposed so B fragments are contiguous
// 32-byte per-lane reads). 8 wmma per 32-wide K step, 12 b128 loads.
// acc[mi][ni]: mi in {0,1} (rows row0+mi*16), ni in 0..3 (cols col0+ni*16).
// ---------------------------------------------------------------------------
template <typename EPI>
__device__ __forceinline__ void gemm32x64(const bf16_t* __restrict__ A, int lda,
                                          const bf16_t* __restrict__ Bt, int ldbt,
                                          int row0, int col0, int K, int lane,
                                          EPI&& epilogue) {
  const int m    = lane & 15;
  const int half = lane >> 4;
  f32x8 acc[2][4] = {};
  const bf16_t* ap0 = A + (size_t)(row0 + m) * lda + half * 16;
  const bf16_t* ap1 = ap0 + (size_t)16 * lda;
  const bf16_t* bp  = Bt + (size_t)(col0 + m) * ldbt + half * 16;
  const size_t bstep = (size_t)16 * ldbt;
  for (int kb = 0; kb < K; kb += 32) {
    bf16x16 a0 = cat8(*(const bf16x8*)(ap0 + kb), *(const bf16x8*)(ap0 + kb + 8));
    bf16x16 a1 = cat8(*(const bf16x8*)(ap1 + kb), *(const bf16x8*)(ap1 + kb + 8));
    bf16x16 b0 = *(const bf16x16*)(bp + kb);              // 32B-aligned
    bf16x16 b1 = *(const bf16x16*)(bp + bstep + kb);
    bf16x16 b2 = *(const bf16x16*)(bp + 2 * bstep + kb);
    bf16x16 b3 = *(const bf16x16*)(bp + 3 * bstep + kb);
    acc[0][0] = wmma_bf16(a0, b0, acc[0][0]);
    acc[0][1] = wmma_bf16(a0, b1, acc[0][1]);
    acc[0][2] = wmma_bf16(a0, b2, acc[0][2]);
    acc[0][3] = wmma_bf16(a0, b3, acc[0][3]);
    acc[1][0] = wmma_bf16(a1, b0, acc[1][0]);
    acc[1][1] = wmma_bf16(a1, b1, acc[1][1]);
    acc[1][2] = wmma_bf16(a1, b2, acc[1][2]);
    acc[1][3] = wmma_bf16(a1, b3, acc[1][3]);
  }
  epilogue(acc);
}

// qkv = xb @ W_attn + b_attn ; scatter into Q (scaled), K [B,H,T,Dh], V^T [B,H,Dh,T]
__global__ __launch_bounds__(128)
void qkv_kernel(const bf16_t* __restrict__ xb, const bf16_t* __restrict__ WattnT,
                const float* __restrict__ b_attn,
                bf16_t* __restrict__ Qs, bf16_t* __restrict__ Ks, bf16_t* __restrict__ Vt) {
  const int lane = threadIdx.x & 31;
  const int wave = threadIdx.x >> 5;
  const int col0 = (blockIdx.x * 4 + wave) * 64;   // 36 wave-columns over 2304
  const int row0 = blockIdx.y * 32;                // 256 row-blocks over 8192
  const int n = lane & 15, half = lane >> 4;

  gemm32x64(xb, C_EMB, WattnT, C_EMB, row0, col0, C_EMB, lane,
    [&](f32x8 (&acc)[2][4]) {
#pragma unroll
      for (int ni = 0; ni < 4; ++ni) {
        const int c = col0 + ni * 16 + n;          // [0, 2304)
        const float bias = b_attn[c];
        const int sec = (c >= 2 * C_EMB) ? 2 : (c >= C_EMB ? 1 : 0);
        const int cc  = c - sec * C_EMB;
        const int h = cc >> 6, d = cc & 63;
#pragma unroll
        for (int mi = 0; mi < 2; ++mi) {
#pragma unroll
          for (int v = 0; v < 8; ++v) {
            int r = row0 + mi * 16 + half * 8 + v; // token in [0, B*T)
            int b = r >> 12, t = r & (T_SEQ - 1);
            float val = acc[mi][ni][v] + bias;
            size_t bh = (size_t)(b * NH + h);
            if (sec == 0)      Qs[(bh * T_SEQ + t) * DH + d] = (bf16_t)(val * 0.125f);
            else if (sec == 1) Ks[(bh * T_SEQ + t) * DH + d] = (bf16_t)val;
            else               Vt[(bh * DH + d) * T_SEQ + t] = (bf16_t)val;
          }
        }
      }
    });
}

// out = Y @ W_proj + b_proj  (f32 output)
__global__ __launch_bounds__(128)
void proj_kernel(const bf16_t* __restrict__ Y, const bf16_t* __restrict__ WprojT,
                 const float* __restrict__ b_proj, float* __restrict__ out) {
  const int lane = threadIdx.x & 31;
  const int wave = threadIdx.x >> 5;
  const int col0 = (blockIdx.x * 4 + wave) * 64;   // 12 wave-columns over 768
  const int row0 = blockIdx.y * 32;
  const int n = lane & 15, half = lane >> 4;

  gemm32x64(Y, C_EMB, WprojT, C_EMB, row0, col0, C_EMB, lane,
    [&](f32x8 (&acc)[2][4]) {
#pragma unroll
      for (int ni = 0; ni < 4; ++ni) {
        const int c = col0 + ni * 16 + n;
        const float bias = b_proj[c];
#pragma unroll
        for (int mi = 0; mi < 2; ++mi) {
#pragma unroll
          for (int v = 0; v < 8; ++v) {
            int r = row0 + mi * 16 + half * 8 + v;
            out[(size_t)r * C_EMB + c] = acc[mi][ni][v] + bias;
          }
        }
      }
    });
}

// Flash attention: 4 waves/block, each wave owns 16 query rows; 32 keys/step.
__global__ __launch_bounds__(128)
void attn_kernel(const bf16_t* __restrict__ Qs, const bf16_t* __restrict__ Ks,
                 const bf16_t* __restrict__ Vt, bf16_t* __restrict__ Y) {
  __shared__ bf16_t Plds[4][16 * 32];
  const int lane = threadIdx.x & 31;
  const int wave = threadIdx.x >> 5;
  const int bh = blockIdx.y;                  // B*H = 24
  const int qbase = blockIdx.x * 64 + wave * 16;
  const int m = lane & 15, half = lane >> 4;

  const bf16_t* Qh = Qs + (size_t)bh * T_SEQ * DH;  // [T,64]
  const bf16_t* Kh = Ks + (size_t)bh * T_SEQ * DH;  // [T,64]
  const bf16_t* Vh = Vt + (size_t)bh * DH * T_SEQ;  // [64,T]

  // Q A-fragments for K-dim 0..31 and 32..63
  const bf16_t* qrow = Qh + (size_t)(qbase + m) * DH + half * 16;
  bf16x16 aq0 = cat8(*(const bf16x8*)(qrow),      *(const bf16x8*)(qrow + 8));
  bf16x16 aq1 = cat8(*(const bf16x8*)(qrow + 32), *(const bf16x8*)(qrow + 40));

  float rmax[8], rsum[8];
  f32x8 o0 = {}, o1 = {}, o2 = {}, o3 = {};
#pragma unroll
  for (int v = 0; v < 8; ++v) { rmax[v] = -3.0e38f; rsum[v] = 0.0f; }

  bf16_t* P = Plds[wave];
  const int qmax = qbase + 15;
  for (int jb = 0; jb <= qmax; jb += 32) {
    // ---- S = Q K^T  (B operand: rows of K are the transposed-B layout) ----
    const bf16_t* k0p = Kh + (size_t)(jb + m) * DH + half * 16;
    const bf16_t* k1p = Kh + (size_t)(jb + 16 + m) * DH + half * 16;
    __builtin_prefetch(k0p + 32 * DH, 0, 3);  // next key tile
    bf16x16 bk00 = cat8(*(const bf16x8*)(k0p),      *(const bf16x8*)(k0p + 8));
    bf16x16 bk01 = cat8(*(const bf16x8*)(k0p + 32), *(const bf16x8*)(k0p + 40));
    bf16x16 bk10 = cat8(*(const bf16x8*)(k1p),      *(const bf16x8*)(k1p + 8));
    bf16x16 bk11 = cat8(*(const bf16x8*)(k1p + 32), *(const bf16x8*)(k1p + 40));
    f32x8 s0 = {}, s1 = {};
    s0 = wmma_bf16(aq0, bk00, s0);
    s0 = wmma_bf16(aq1, bk01, s0);
    s1 = wmma_bf16(aq0, bk10, s1);
    s1 = wmma_bf16(aq1, bk11, s1);

    // ---- causal mask + online softmax (rows live in 16-lane halves) ----
    float p0[8], p1[8], alpha[8];
#pragma unroll
    for (int v = 0; v < 8; ++v) {
      int q  = qbase + half * 8 + v;
      int n0 = jb + m;
      float s0v = (n0 > q)      ? -3.0e38f : s0[v];
      float s1v = (n0 + 16 > q) ? -3.0e38f : s1[v];
      float mx = fmaxf(s0v, s1v);
      mx = fmaxf(mx, __shfl_xor(mx, 1, 32));
      mx = fmaxf(mx, __shfl_xor(mx, 2, 32));
      mx = fmaxf(mx, __shfl_xor(mx, 4, 32));
      mx = fmaxf(mx, __shfl_xor(mx, 8, 32));
      float mnew = fmaxf(rmax[v], mx);
      alpha[v] = __expf(rmax[v] - mnew);
      rmax[v] = mnew;
      float e0 = __expf(s0v - mnew);
      float e1 = __expf(s1v - mnew);
      float rs = e0 + e1;
      rs += __shfl_xor(rs, 1, 32);
      rs += __shfl_xor(rs, 2, 32);
      rs += __shfl_xor(rs, 4, 32);
      rs += __shfl_xor(rs, 8, 32);
      rsum[v] = rsum[v] * alpha[v] + rs;
      p0[v] = e0; p1[v] = e1;
    }

    // rescale O, stage P (C-layout -> LDS) per-wave; DS is in-order per wave
#pragma unroll
    for (int v = 0; v < 8; ++v) {
      o0[v] *= alpha[v]; o1[v] *= alpha[v]; o2[v] *= alpha[v]; o3[v] *= alpha[v];
      int row = half * 8 + v;
      P[row * 32 + m]      = (bf16_t)p0[v];
      P[row * 32 + 16 + m] = (bf16_t)p1[v];
    }
    asm volatile("" ::: "memory");
    const bf16_t* pr = P + m * 32 + half * 16;
    bf16x16 ap = cat8(*(const bf16x8*)(pr), *(const bf16x8*)(pr + 8));
    asm volatile("" ::: "memory");

    // ---- O += P V  (B operand: rows of V^T) ----
    const bf16_t* vp = Vh + (size_t)m * T_SEQ + jb + half * 16;
    __builtin_prefetch(vp + 32, 0, 3);
    bf16x16 bv0 = cat8(*(const bf16x8*)(vp),              *(const bf16x8*)(vp + 8));
    bf16x16 bv1 = cat8(*(const bf16x8*)(vp + 16 * T_SEQ), *(const bf16x8*)(vp + 16 * T_SEQ + 8));
    bf16x16 bv2 = cat8(*(const bf16x8*)(vp + 32 * T_SEQ), *(const bf16x8*)(vp + 32 * T_SEQ + 8));
    bf16x16 bv3 = cat8(*(const bf16x8*)(vp + 48 * T_SEQ), *(const bf16x8*)(vp + 48 * T_SEQ + 8));
    o0 = wmma_bf16(ap, bv0, o0);
    o1 = wmma_bf16(ap, bv1, o1);
    o2 = wmma_bf16(ap, bv2, o2);
    o3 = wmma_bf16(ap, bv3, o3);
  }

  // ---- finalize: O /= rowsum, write Y [B*T, C] in head-interleaved columns ----
  const int b = bh / NH, h = bh % NH;
#pragma unroll
  for (int v = 0; v < 8; ++v) {
    float inv = 1.0f / rsum[v];
    size_t tok = (size_t)b * T_SEQ + qbase + half * 8 + v;
    bf16_t* yr = Y + tok * C_EMB + h * DH + m;
    yr[0]  = (bf16_t)(o0[v] * inv);
    yr[16] = (bf16_t)(o1[v] * inv);
    yr[32] = (bf16_t)(o2[v] * inv);
    yr[48] = (bf16_t)(o3[v] * inv);
  }
}

extern "C" void kernel_launch(void* const* d_in, const int* in_sizes, int n_in,
                              void* d_out, int out_size, void* d_ws, size_t ws_size,
                              hipStream_t stream) {
  (void)in_sizes; (void)n_in; (void)out_size; (void)ws_size;
  const float* x      = (const float*)d_in[0];
  const float* W_attn = (const float*)d_in[1];
  const float* b_attn = (const float*)d_in[2];
  const float* W_proj = (const float*)d_in[3];
  const float* b_proj = (const float*)d_in[4];
  float* out = (float*)d_out;

  char* ws = (char*)d_ws;
  bf16_t* xb     = (bf16_t*)ws; ws += (size_t)NB * T_SEQ * C_EMB * 2;   // 12.6 MB
  bf16_t* WattnT = (bf16_t*)ws; ws += (size_t)C3 * C_EMB * 2;           //  3.5 MB
  bf16_t* WprojT = (bf16_t*)ws; ws += (size_t)C_EMB * C_EMB * 2;        //  1.2 MB
  bf16_t* Qs     = (bf16_t*)ws; ws += (size_t)NB * NH * T_SEQ * DH * 2; // 12.6 MB
  bf16_t* Ks     = (bf16_t*)ws; ws += (size_t)NB * NH * T_SEQ * DH * 2; // 12.6 MB
  bf16_t* Vt     = (bf16_t*)ws; ws += (size_t)NB * NH * T_SEQ * DH * 2; // 12.6 MB
  bf16_t* Yb     = (bf16_t*)ws;                                         // 12.6 MB

  const int nx = NB * T_SEQ * C_EMB;
  cvt_bf16_kernel<<<(nx + 255) / 256, 256, 0, stream>>>(x, xb, nx);
  transpose_cvt_kernel<<<(C3 * C_EMB + 255) / 256, 256, 0, stream>>>(W_attn, WattnT, C_EMB, C3);
  transpose_cvt_kernel<<<(C_EMB * C_EMB + 255) / 256, 256, 0, stream>>>(W_proj, WprojT, C_EMB, C_EMB);

  // 32 rows x 256 cols per block (4 waves x 64 cols each)
  qkv_kernel <<<dim3(C3 / 256,    NB * T_SEQ / 32), 128, 0, stream>>>(xb, WattnT, b_attn, Qs, Ks, Vt);
  attn_kernel<<<dim3(T_SEQ / 64,  NB * NH),         128, 0, stream>>>(Qs, Ks, Vt, Yb);
  proj_kernel<<<dim3(C_EMB / 256, NB * T_SEQ / 32), 128, 0, stream>>>(Yb, WprojT, b_proj, out);
}
